// ReCorr_30880814858290
// MI455X (gfx1250) — matrix operand order repaired and checked by reading, not compiled, not run on hardware
//
#include <hip/hip_runtime.h>
#include <math.h>

typedef __attribute__((ext_vector_type(16))) _Float16 v16h;
typedef __attribute__((ext_vector_type(8)))  float    v8f;

#define ACT_NONE 0
#define ACT_RELU 1
#define ACT_SIG  2
#define ACT_TANH 3

static inline int cdiv(int a, int b) { return (a + b - 1) / b; }

__device__ __forceinline__ int imin_(int a, int b) { return a < b ? a : b; }
__device__ __forceinline__ int imax_(int a, int b) { return a > b ? a : b; }

// ---------------------------------------------------------------- elementwise
__global__ void k_zero(float* p, int n) {
  int i = blockIdx.x * blockDim.x + threadIdx.x;
  if (i < n) p[i] = 0.f;
}
__global__ void k_cast16(_Float16* d, const float* s, int n) {
  int i = blockIdx.x * blockDim.x + threadIdx.x;
  if (i < n) d[i] = (_Float16)s[i];
}
__global__ void k_copy16(_Float16* d, const _Float16* s, int n) {
  int i = blockIdx.x * blockDim.x + threadIdx.x;
  if (i < n) d[i] = s[i];
}
// hid = tanh(fmap1[0:16]); ctx = relu(fmap1[16:32]) -> two f16 concat slots
__global__ void k_hidctx(const float* f1, float* hid, _Float16* hid16,
                         _Float16* ctxA, _Float16* ctxB, int V) {
  int i = blockIdx.x * blockDim.x + threadIdx.x;
  if (i >= 16 * V) return;
  int c = i / V, v = i - c * V;
  float hv = tanhf(f1[c * V + v]);
  hid[i] = hv; hid16[i] = (_Float16)hv;
  float cv = fmaxf(f1[(16 + c) * V + v], 0.f);
  ctxA[i] = (_Float16)cv; ctxB[i] = (_Float16)cv;
}
// qin[0:16V] = f16(r * hid)
__global__ void k_rh(_Float16* qin, const float* r, const float* hid, int n) {
  int i = blockIdx.x * blockDim.x + threadIdx.x;
  if (i < n) qin[i] = (_Float16)(r[i] * hid[i]);
}
// hid = (1-z)*hid + z*q
__global__ void k_gru(float* hid, _Float16* hid16, const float* z, const float* q, int n) {
  int i = blockIdx.x * blockDim.x + threadIdx.x;
  if (i >= n) return;
  float zz = z[i];
  float hv = (1.f - zz) * hid[i] + zz * q[i];
  hid[i] = hv; hid16[i] = (_Float16)hv;
}
__global__ void k_addflow(float* flow, _Float16* flow16, const float* df, int n) {
  int i = blockIdx.x * blockDim.x + threadIdx.x;
  if (i >= n) return;
  float v = flow[i] + df[i];
  flow[i] = v; flow16[i] = (_Float16)v;
}

// ------------------------------------------------------------------- trilerp
__device__ float trilerp(const float* vol, int D, int H, int W,
                         float z, float y, float x) {
  float z0 = floorf(z), y0 = floorf(y), x0 = floorf(x);
  float fz = z - z0, fy = y - y0, fx = x - x0;
  int z0i = (int)z0, y0i = (int)y0, x0i = (int)x0;
  float acc = 0.f;
#pragma unroll
  for (int dz = 0; dz < 2; dz++)
#pragma unroll
    for (int dy = 0; dy < 2; dy++)
#pragma unroll
      for (int dx = 0; dx < 2; dx++) {
        int zi = z0i + dz, yi = y0i + dy, xi = x0i + dx;
        bool valid = (zi >= 0) && (zi < D) && (yi >= 0) && (yi < H) && (xi >= 0) && (xi < W);
        float wz = dz ? fz : 1.f - fz;
        float wy = dy ? fy : 1.f - fy;
        float wx = dx ? fx : 1.f - fx;
        int zc = imin_(imax_(zi, 0), D - 1);
        int yc = imin_(imax_(yi, 0), H - 1);
        int xc = imin_(imax_(xi, 0), W - 1);
        acc += vol[(zc * H + yc) * W + xc] * (valid ? wz * wy * wx : 0.f);
      }
  return acc;
}

// ---------------------------------------------------------- correlation search
// one thread per (offset o in 0..124, voxel); writes f16 corr [125][V]
__global__ void k_corr(const float* __restrict__ f1, const float* __restrict__ f2,
                       const float* __restrict__ flow, _Float16* __restrict__ corr16,
                       int D, int H, int W, int V, int C, float inv) {
  int t = blockIdx.x * blockDim.x + threadIdx.x;
  if (t >= 125 * V) return;
  int vox = t % V, o = t / V;
  int z = vox / (H * W); int r = vox - z * H * W; int y = r / W; int x = r - y * W;
  float cz = (float)z + flow[0 * V + vox] + (float)(o / 25 - 2);
  float cy = (float)y + flow[1 * V + vox] + (float)((o / 5) % 5 - 2);
  float cx = (float)x + flow[2 * V + vox] + (float)(o % 5 - 2);
  float z0 = floorf(cz), y0 = floorf(cy), x0 = floorf(cx);
  float fz = cz - z0, fy = cy - y0, fx = cx - x0;
  int z0i = (int)z0, y0i = (int)y0, x0i = (int)x0;
  int idx8[8]; float w8[8]; int u = 0;
#pragma unroll
  for (int dz = 0; dz < 2; dz++)
#pragma unroll
    for (int dy = 0; dy < 2; dy++)
#pragma unroll
      for (int dx = 0; dx < 2; dx++) {
        int zi = z0i + dz, yi = y0i + dy, xi = x0i + dx;
        bool valid = (zi >= 0) && (zi < D) && (yi >= 0) && (yi < H) && (xi >= 0) && (xi < W);
        float wz = dz ? fz : 1.f - fz;
        float wy = dy ? fy : 1.f - fy;
        float wx = dx ? fx : 1.f - fx;
        int zc = imin_(imax_(zi, 0), D - 1);
        int yc = imin_(imax_(yi, 0), H - 1);
        int xc = imin_(imax_(xi, 0), W - 1);
        idx8[u] = (zc * H + yc) * W + xc;
        w8[u] = valid ? wz * wy * wx : 0.f;
        u++;
      }
  float acc = 0.f;
  for (int c = 0; c < C; c++) {
    const float* p2 = f2 + c * V;
    float s = 0.f;
#pragma unroll
    for (int u2 = 0; u2 < 8; u2++) s += p2[idx8[u2]] * w8[u2];
    acc += f1[c * V + vox] * s;
  }
  corr16[o * V + vox] = (_Float16)(acc * inv);
}

// ----------------------------------------------------- weight prep (f32->f16 pad)
// w layout [Cout][Cin][K3] (Ktot = Cin*K3 contiguous per cout row)
__global__ void k_prepw(const float* __restrict__ w, _Float16* __restrict__ wpad,
                        int Cout, int Ktot, int Kpad, int CoutPad) {
  int i = blockIdx.x * blockDim.x + threadIdx.x;
  if (i >= CoutPad * Kpad) return;
  int row = i / Kpad, k = i - row * Kpad;
  float v = (row < Cout && k < Ktot) ? w[row * Ktot + k] : 0.f;
  wpad[i] = (_Float16)v;
}

// ----------------------------------------------------- implicit-GEMM conv (WMMA)
// in16: [Cin][D*H*W] f16.  wpad: [CoutPad][Kpad] f16 (K = cin*K3 + tap, padded w/ 0).
// Each wave owns one 16-voxel M tile and ALL NT cout tiles: the expensive
// im2col A-gather is amortized over NT back-to-back v_wmma issues.
// NT = CoutPad/16 (compile-time), K1 = (kernel is 1x1x1).
template <int NT, bool K1>
__global__ void k_conv_wmma(const _Float16* __restrict__ in16,
                            const _Float16* __restrict__ wpad,
                            const float* __restrict__ bias,
                            float* __restrict__ out32, _Float16* __restrict__ out16,
                            int Ktot, int Kpad, int Cout,
                            int D, int H, int W, int V, int act) {
  int lane = threadIdx.x & 31;
  int wave = threadIdx.x >> 5;
  int tile_m = blockIdx.x * 4 + wave;
  if (tile_m * 16 >= V) return;            // wave-uniform exit; EXEC stays full
  int half = lane >> 4;
  int mrow = lane & 15;

  int vox = tile_m * 16 + mrow;            // A-operand row for this lane
  bool vok = vox < V;
  int zi = 0, yi = 0, xi = 0;
  if (vok) { zi = vox / (H * W); int rr = vox - zi * H * W; yi = rr / W; xi = rr - yi * W; }

  // B-operand: lane holds weight column cout = nt*16 + mrow, contiguous K slice.
  const _Float16* wbase = wpad + mrow * Kpad + half * 16;

  v8f acc[NT];
#pragma unroll
  for (int t = 0; t < NT; t++) acc[t] = (v8f){0.f, 0.f, 0.f, 0.f, 0.f, 0.f, 0.f, 0.f};

  for (int k0 = 0; k0 < Kpad; k0 += 32) {
    v16h a;
#pragma unroll
    for (int e = 0; e < 16; e++) {
      // A k order used here: K = half*16 + e (matches contiguous B)
      int k = k0 + (half << 4) + e;
      _Float16 val = (_Float16)0.f;
      if (vok && k < Ktot) {
        if (K1) {
          val = in16[k * V + vox];         // coalesced across lanes
        } else {
          int cin = k / 27, tap = k - cin * 27;   // constant-divisor magic mul
          int zz = zi + tap / 9 - 1;
          int t2 = tap % 9;
          int yy = yi + t2 / 3 - 1;
          int xx = xi + t2 % 3 - 1;
          if (zz >= 0 && zz < D && yy >= 0 && yy < H && xx >= 0 && xx < W)
            val = in16[cin * V + (zz * H + yy) * W + xx];
        }
      }
      a[e] = val;
    }
#pragma unroll
    for (int t = 0; t < NT; t++) {
      const v16h b = *(const v16h*)(wbase + t * 16 * Kpad + k0);
      acc[t] = __builtin_amdgcn_wmma_f32_16x16x32_f16(false, a, false, b,
                                                      (short)0, acc[t], false, false);
    }
  }

  // D layout: element i -> M = half*8 + i, N = lane&15
#pragma unroll
  for (int t = 0; t < NT; t++) {
    int co = t * 16 + mrow;
    if (co < Cout) {
      float bb = bias ? bias[co] : 0.f;
#pragma unroll
      for (int i = 0; i < 8; i++) {
        int mv = tile_m * 16 + half * 8 + i;
        if (mv < V) {
          float v = acc[t][i] + bb;
          if (act == ACT_RELU) v = fmaxf(v, 0.f);
          else if (act == ACT_SIG) v = 1.f / (1.f + expf(-v));
          else if (act == ACT_TANH) v = tanhf(v);
          if (out32) out32[co * V + mv] = v;
          if (out16) out16[co * V + mv] = (_Float16)v;
        }
      }
    }
  }
}

// --------------------------------------------------------------------- upflow
__global__ void k_upflow(const float* src, int D, int H, int W,
                         float* dst, int ND, int NH, int NW) {
  int NV = ND * NH * NW;
  int t = blockIdx.x * blockDim.x + threadIdx.x;
  if (t >= 3 * NV) return;
  int c = t / NV, v = t - c * NV;
  int iz = v / (NH * NW); int rr = v - iz * NH * NW; int iy = rr / NW; int ix = rr - iy * NW;
  float z = iz * ((float)(D - 1) / (float)(ND - 1));
  float y = iy * ((float)(H - 1) / (float)(NH - 1));
  float x = ix * ((float)(W - 1) / (float)(NW - 1));
  dst[t] = 2.f * trilerp(src + c * D * H * W, D, H, W, z, y, x);
}

// ----------------------------------------------------------------------- warp
__global__ void k_warp(const float* img, const float* disp, float* out,
                       int D, int H, int W) {
  int V = D * H * W;
  int t = blockIdx.x * blockDim.x + threadIdx.x;
  if (t >= V) return;
  int iz = t / (H * W); int rr = t - iz * H * W; int iy = rr / W; int ix = rr - iy * W;
  out[t] = trilerp(img, D, H, W,
                   iz + disp[0 * V + t], iy + disp[1 * V + t], ix + disp[2 * V + t]);
}

// =========================================================== host orchestration
struct StageBufs {
  float *hid, *z, *r, *q, *df;
  _Float16 *hid16, *flow16, *corr16, *c1, *f1, *cat32, *hx, *qin, *h1, *wpad;
};

static void launch_conv(hipStream_t st, _Float16* wpad,
                        const float* w, const float* b,
                        const _Float16* in16, float* out32, _Float16* out16,
                        int Cout, int Cin, int K3, int D, int H, int W, int act) {
  int V = D * H * W;
  int Ktot = Cin * K3;
  int Kpad = cdiv(Ktot, 32) * 32;
  int CoutPad = cdiv(Cout, 16) * 16;
  int nprep = CoutPad * Kpad;
  k_prepw<<<cdiv(nprep, 256), 256, 0, st>>>(w, wpad, Cout, Ktot, Kpad, CoutPad);
  dim3 grid(cdiv(cdiv(V, 16), 4));
  int NT = CoutPad / 16;
  if (K3 == 1) {
    if (NT == 1)      k_conv_wmma<1, true><<<grid, 128, 0, st>>>(in16, wpad, b, out32, out16, Ktot, Kpad, Cout, D, H, W, V, act);
    else if (NT == 2) k_conv_wmma<2, true><<<grid, 128, 0, st>>>(in16, wpad, b, out32, out16, Ktot, Kpad, Cout, D, H, W, V, act);
    else              k_conv_wmma<3, true><<<grid, 128, 0, st>>>(in16, wpad, b, out32, out16, Ktot, Kpad, Cout, D, H, W, V, act);
  } else {
    if (NT == 1)      k_conv_wmma<1, false><<<grid, 128, 0, st>>>(in16, wpad, b, out32, out16, Ktot, Kpad, Cout, D, H, W, V, act);
    else if (NT == 2) k_conv_wmma<2, false><<<grid, 128, 0, st>>>(in16, wpad, b, out32, out16, Ktot, Kpad, Cout, D, H, W, V, act);
    else              k_conv_wmma<3, false><<<grid, 128, 0, st>>>(in16, wpad, b, out32, out16, Ktot, Kpad, Cout, D, H, W, V, act);
  }
}

// Assumes harness flattens params in setup_inputs() insertion order:
// blocks 16x,8x,4x,2x each [corr1,corr2,flowc1,flowc2,motion,convz,convr,convq,fh1,fh2] x {w,b}
static void run_stage(hipStream_t st, const StageBufs& S, void* const* d_in, int pbase,
                      const float* f1, const float* f2, float* flow,
                      int D, int H, int W, int iters) {
  int V = D * H * W;
  auto Wp = [&](int j) { return (const float*)d_in[pbase + 2 * j]; };
  auto Bp = [&](int j) { return (const float*)d_in[pbase + 2 * j + 1]; };
  int T = 256;

  // hid/ctx init; ctx written straight into the two 51-channel concat buffers
  k_hidctx<<<cdiv(16 * V, T), T, 0, st>>>(f1, S.hid, S.hid16, S.hx + 16 * V, S.qin + 16 * V, V);
  k_cast16<<<cdiv(3 * V, T), T, 0, st>>>(S.flow16, flow, 3 * V);

  for (int it = 0; it < iters; ++it) {
    k_corr<<<cdiv(125 * V, 128), 128, 0, st>>>(f1, f2, flow, S.corr16, D, H, W, V, 32,
                                               0.17677669529663689f);
    launch_conv(st, S.wpad, Wp(0), Bp(0), S.corr16, nullptr, S.c1, 48, 125, 1, D, H, W, ACT_RELU);
    launch_conv(st, S.wpad, Wp(1), Bp(1), S.c1, nullptr, S.cat32, 16, 48, 27, D, H, W, ACT_RELU);
    launch_conv(st, S.wpad, Wp(2), Bp(2), S.flow16, nullptr, S.f1, 16, 3, 27, D, H, W, ACT_RELU);
    launch_conv(st, S.wpad, Wp(3), Bp(3), S.f1, nullptr, S.cat32 + 16 * V, 16, 16, 27, D, H, W, ACT_RELU);
    launch_conv(st, S.wpad, Wp(4), Bp(4), S.cat32, nullptr, S.hx + 32 * V, 16, 32, 27, D, H, W, ACT_RELU);
    // assemble hx = [hid|ctx|m|flow], qin = [r*hid|ctx|m|flow]
    k_copy16<<<cdiv(16 * V, T), T, 0, st>>>(S.qin + 32 * V, S.hx + 32 * V, 16 * V);
    k_copy16<<<cdiv(3 * V, T), T, 0, st>>>(S.hx + 48 * V, S.flow16, 3 * V);
    k_copy16<<<cdiv(3 * V, T), T, 0, st>>>(S.qin + 48 * V, S.flow16, 3 * V);
    k_copy16<<<cdiv(16 * V, T), T, 0, st>>>(S.hx, S.hid16, 16 * V);
    launch_conv(st, S.wpad, Wp(5), Bp(5), S.hx, S.z, nullptr, 16, 51, 27, D, H, W, ACT_SIG);
    launch_conv(st, S.wpad, Wp(6), Bp(6), S.hx, S.r, nullptr, 16, 51, 27, D, H, W, ACT_SIG);
    k_rh<<<cdiv(16 * V, T), T, 0, st>>>(S.qin, S.r, S.hid, 16 * V);
    launch_conv(st, S.wpad, Wp(7), Bp(7), S.qin, S.q, nullptr, 16, 51, 27, D, H, W, ACT_TANH);
    k_gru<<<cdiv(16 * V, T), T, 0, st>>>(S.hid, S.hid16, S.z, S.q, 16 * V);
    launch_conv(st, S.wpad, Wp(8), Bp(8), S.hid16, nullptr, S.h1, 32, 16, 27, D, H, W, ACT_RELU);
    launch_conv(st, S.wpad, Wp(9), Bp(9), S.h1, S.df, nullptr, 3, 32, 27, D, H, W, ACT_NONE);
    k_addflow<<<cdiv(3 * V, T), T, 0, st>>>(flow, S.flow16, S.df, 3 * V);
  }
}

extern "C" void kernel_launch(void* const* d_in, const int* in_sizes, int n_in,
                              void* d_out, int out_size, void* d_ws, size_t ws_size,
                              hipStream_t stream) {
  (void)in_sizes; (void)n_in; (void)out_size; (void)ws_size;
  const int V2 = 48 * 48 * 48;      // largest recurrent stage
  const int V1 = 96 * 96 * 96;      // full resolution

  char* p = (char*)d_ws;
  auto alloc = [&](size_t bytes) -> void* {
    void* r = (void*)p; p += (bytes + 255) & ~(size_t)255; return r;
  };

  StageBufs S;
  S.hid    = (float*)alloc((size_t)16 * V2 * 4);
  S.z      = (float*)alloc((size_t)16 * V2 * 4);
  S.r      = (float*)alloc((size_t)16 * V2 * 4);
  S.q      = (float*)alloc((size_t)16 * V2 * 4);
  S.df     = (float*)alloc((size_t)3 * V2 * 4);
  S.hid16  = (_Float16*)alloc((size_t)16 * V2 * 2);
  S.flow16 = (_Float16*)alloc((size_t)3 * V2 * 2);
  S.corr16 = (_Float16*)alloc((size_t)125 * V2 * 2);
  S.c1     = (_Float16*)alloc((size_t)48 * V2 * 2);
  S.f1     = (_Float16*)alloc((size_t)16 * V2 * 2);
  S.cat32  = (_Float16*)alloc((size_t)32 * V2 * 2);
  S.hx     = (_Float16*)alloc((size_t)51 * V2 * 2);
  S.qin    = (_Float16*)alloc((size_t)51 * V2 * 2);
  S.h1     = (_Float16*)alloc((size_t)32 * V2 * 2);
  S.wpad   = (_Float16*)alloc((size_t)64 * 1024);
  float* flowA = (float*)alloc((size_t)3 * V1 * 4);
  float* flowB = (float*)alloc((size_t)3 * V1 * 4);
  _Float16* cat35 = (_Float16*)alloc((size_t)35 * V1 * 2);

  const float* fmap1_16x = (const float*)d_in[0];
  const float* fmap2_16x = (const float*)d_in[1];
  const float* fmap1_8x  = (const float*)d_in[2];
  const float* fmap2_8x  = (const float*)d_in[3];
  const float* fmap1_4x  = (const float*)d_in[4];
  const float* fmap2_4x  = (const float*)d_in[5];
  const float* fmap1_2x  = (const float*)d_in[6];
  const float* fmap2_2x  = (const float*)d_in[7];
  const float* fmap1_1x  = (const float*)d_in[8];
  const float* fmap2_1x  = (const float*)d_in[9];
  const float* image2    = (const float*)d_in[10];
  const int T = 256;

  // 16x stage (6^3), 3 iters
  k_zero<<<cdiv(3 * 216, T), T, 0, stream>>>(flowA, 3 * 216);
  run_stage(stream, S, d_in, 11, fmap1_16x, fmap2_16x, flowA, 6, 6, 6, 3);
  k_upflow<<<cdiv(3 * 1728, T), T, 0, stream>>>(flowA, 6, 6, 6, flowB, 12, 12, 12);
  // 8x stage (12^3), 3 iters
  run_stage(stream, S, d_in, 31, fmap1_8x, fmap2_8x, flowB, 12, 12, 12, 3);
  k_upflow<<<cdiv(3 * 13824, T), T, 0, stream>>>(flowB, 12, 12, 12, flowA, 24, 24, 24);
  // 4x stage (24^3), 2 iters
  run_stage(stream, S, d_in, 51, fmap1_4x, fmap2_4x, flowA, 24, 24, 24, 2);
  k_upflow<<<cdiv(3 * V2, T), T, 0, stream>>>(flowA, 24, 24, 24, flowB, 48, 48, 48);
  // 2x stage (48^3), 2 iters
  run_stage(stream, S, d_in, 71, fmap1_2x, fmap2_2x, flowB, 48, 48, 48, 2);
  k_upflow<<<cdiv(3 * V1, T), T, 0, stream>>>(flowB, 48, 48, 48, flowA, 96, 96, 96);

  // final: disp = conv3([fmap1_1x | fmap2_1x | flow], flow_head); moved = warp(image2)
  k_cast16<<<cdiv(16 * V1, T), T, 0, stream>>>(cat35, fmap1_1x, 16 * V1);
  k_cast16<<<cdiv(16 * V1, T), T, 0, stream>>>(cat35 + (size_t)16 * V1, fmap2_1x, 16 * V1);
  k_cast16<<<cdiv(3 * V1, T), T, 0, stream>>>(cat35 + (size_t)32 * V1, flowA, 3 * V1);
  launch_conv(stream, S.wpad, (const float*)d_in[91], (const float*)d_in[92],
              cat35, (float*)d_out, nullptr, 3, 35, 27, 96, 96, 96, ACT_NONE);
  k_warp<<<cdiv(V1, T), T, 0, stream>>>(image2, (const float*)d_out,
                                        (float*)d_out + (size_t)3 * V1, 96, 96, 96);
}
